// GCNEncoder_20495583936893
// MI455X (gfx1250) — compile-verified, matrix operand order checked by reading, not run on hardware
//
#include <hip/hip_runtime.h>

typedef __attribute__((ext_vector_type(16))) __bf16 v16bf;
typedef __attribute__((ext_vector_type(8)))  __bf16 v8bf;
typedef __attribute__((ext_vector_type(8)))  float  v8f;

#define NNODES 1024
#define BATCH  64
#define NTOT   (NNODES * BATCH)      // 65536
#define NEDGE  262144
#define H1DIM  512
#define H2DIM  16
#define H3DIM  4096
#define LATDIM 512
#define NCLS   160
#define K1DIM  (NNODES * H2DIM)          // 16384
#define K2DIM  (NNODES * H2DIM + NCLS)   // 16544 = 32*11*47

// ---------------- bf16 conversion (round-to-nearest-even) ----------------
__device__ __forceinline__ __bf16 f2bf(float f) {
  union { float f; unsigned u; } a; a.f = f;
  unsigned r = a.u + 0x7FFFu + ((a.u >> 16) & 1u);
  union { unsigned short s; __bf16 b; } o; o.s = (unsigned short)(r >> 16);
  return o.b;
}
// pack two floats -> bf16x2 dword (lo = first k, hi = second k)
__device__ __forceinline__ unsigned pack_bf16x2(float lo, float hi) {
  union { float f; unsigned u; } a, b; a.f = lo; b.f = hi;
  unsigned ra = (a.u + 0x7FFFu + ((a.u >> 16) & 1u)) >> 16;
  unsigned rb = (b.u + 0x7FFFu + ((b.u >> 16) & 1u)) >> 16;
  return ra | (rb << 16);
}

// ---------------- utility: zero f32 ----------------
__global__ void k_zero_f32(float* __restrict__ p, int n) {
  for (int i = blockIdx.x * blockDim.x + threadIdx.x; i < n; i += gridDim.x * blockDim.x)
    p[i] = 0.0f;
}

// ---------------- edge scatter: Y[dst] += Xin[src], 8 channels ----------------
// segment_sum(Xin[src], dst) == Adj @ Xin ; linearity folds the 512-wide GCN
// scatter down to an 8-wide one: Adj(X W1) W2 == (Adj^2 X)(W1 W2).
__global__ void k_scatter8(const int* __restrict__ src, const int* __restrict__ dst,
                           const float* __restrict__ Xin, float* __restrict__ Yout) {
  int e = blockIdx.x * blockDim.x + threadIdx.x;
  if (e >= NEDGE) return;
  int s = src[e], d = dst[e];
  const float4* xp = (const float4*)(Xin + (size_t)s * 8);
  float4 a = xp[0], b = xp[1];
  float* yp = Yout + (size_t)d * 8;
  atomicAdd(yp + 0, a.x); atomicAdd(yp + 1, a.y);
  atomicAdd(yp + 2, a.z); atomicAdd(yp + 3, a.w);
  atomicAdd(yp + 4, b.x); atomicAdd(yp + 5, b.y);
  atomicAdd(yp + 6, b.z); atomicAdd(yp + 7, b.w);
}

// ---------------- fused Wg = W_g1 @ W_g2 : [8,512]@[512,16] -> [8,16] ----------------
__global__ void k_fuse_wg(const float* __restrict__ Wg1, const float* __restrict__ Wg2,
                          float* __restrict__ Wg) {
  int t = threadIdx.x;             // 128 threads
  if (t >= 128) return;
  int r = t >> 4, c = t & 15;
  float s = 0.0f;
  for (int j = 0; j < H1DIM; ++j) s += Wg1[r * H1DIM + j] * Wg2[j * H2DIM + c];
  Wg[r * H2DIM + c] = s;
}

// ---------------- x2 = (Adj^2 X) @ Wg, packed into concat([cls, x2]) bf16 ----------------
__global__ void k_build_x2cat(const float* __restrict__ Z, const float* __restrict__ Wg,
                              __bf16* __restrict__ A2) {
  int n = blockIdx.x * blockDim.x + threadIdx.x;
  if (n >= NTOT) return;
  int b = n >> 10, p = n & 1023;
  float z[8];
  const float4* zp = (const float4*)(Z + (size_t)n * 8);
  float4 z0 = zp[0], z1 = zp[1];
  z[0]=z0.x; z[1]=z0.y; z[2]=z0.z; z[3]=z0.w; z[4]=z1.x; z[5]=z1.y; z[6]=z1.z; z[7]=z1.w;
  __bf16* out = A2 + (size_t)b * K2DIM + NCLS + (size_t)p * H2DIM;
  #pragma unroll
  for (int h = 0; h < H2DIM; ++h) {
    float v = 0.0f;
    #pragma unroll
    for (int j = 0; j < 8; ++j) v += z[j] * Wg[j * H2DIM + h];
    out[h] = f2bf(v);
  }
}

__global__ void k_fill_cls(const float* __restrict__ cls, __bf16* __restrict__ A2) {
  int i = blockIdx.x * blockDim.x + threadIdx.x;
  if (i >= BATCH * NCLS) return;
  int b = i / NCLS, c = i % NCLS;
  A2[(size_t)b * K2DIM + c] = f2bf(cls[i]);
}

// ---------------- mix_in = relu(box branch) + relu(label branch), bf16 ----------------
__global__ void k_mixin(const float* __restrict__ X, const float* __restrict__ Wbox,
                        const float* __restrict__ bbox, const float* __restrict__ Wlbl,
                        const float* __restrict__ blbl, __bf16* __restrict__ M) {
  int n = blockIdx.x * blockDim.x + threadIdx.x;
  if (n >= NTOT) return;
  const float* xr = X + (size_t)n * 8;
  float x0 = xr[0];
  float xb[7];
  #pragma unroll
  for (int j = 0; j < 7; ++j) xb[j] = xr[1 + j];
  __bf16* out = M + (size_t)n * H2DIM;
  #pragma unroll
  for (int h = 0; h < H2DIM; ++h) {
    float bx = bbox[h];
    #pragma unroll
    for (int j = 0; j < 7; ++j) bx += xb[j] * Wbox[j * H2DIM + h];
    float lb = blbl[h] + x0 * Wlbl[h];
    float v = (bx > 0.f ? bx : 0.f) + (lb > 0.f ? lb : 0.f);
    out[h] = f2bf(v);
  }
}

// ---------------- bf16 WMMA split-K GEMM: Cacc[64,N] += A[64,K] @ W[K,N] ----------------
// C tile per block: 64 x 128. 8 waves = 4 M-subtiles x 2 N-subtile groups; each wave
// computes 4 WMMA tiles (n = nBase + {0,32,64,96}), reusing one A fragment 4x.
// All five fragment loads are issued before the four WMMAs so a single DS wait covers
// them and the WMMAs run back-to-back on the matrix pipe.
// Fragment layouts per CDNA5 ISA 7.12.2 (wave32):
//   A 16x32 bf16 : lane (l&15)=row; elems 0..7 -> K = 8*(l>>4)+i ; 8..15 -> K = 16+8*(l>>4)+(i-8)
//   B 32x16 bf16 : lane (l&15)=col; same K striping   (LDS B kept [n][k] so frag = 2x b128)
//   C 16x16 f32  : VGPR j -> row j + 8*(l>>4), col = l&15
__device__ __forceinline__ v16bf ldfrag(const __bf16* p) {
  v8bf x = *(const v8bf*)(p);
  v8bf y = *(const v8bf*)(p + 16);
  v16bf r;
  #pragma unroll
  for (int i = 0; i < 8; ++i) { r[i] = x[i]; r[i + 8] = y[i]; }
  return r;
}

__global__ __launch_bounds__(256) void k_gemm_bf16(
    const __bf16* __restrict__ A, int ldA,      // [64, K] bf16 row-major
    const float* __restrict__ W, int N,         // [K, N] f32 row-major (streamed once)
    float* __restrict__ Cacc,                   // [64, N] f32, pre-zeroed
    int K, int kSplit) {
  __shared__ __bf16 lA[64 * 32];                // A tile, [m][k]
  __shared__ __bf16 lB[128 * 32];               // W tile transposed, [n][k]
  const int n0 = blockIdx.x * 128;
  const int kChunk = K / kSplit;                // multiple of 32 by construction
  const int kBeg = blockIdx.y * kChunk;
  const int kEnd = kBeg + kChunk;
  const int t = threadIdx.x;
  const int wave = t >> 5, lane = t & 31;
  const int hl = lane >> 4, lr = lane & 15;
  const int mBase = (wave & 3) * 16;            // 4 M-subtiles
  const int nBase = (wave >> 2) * 16;           // 0 or 16
  v8f acc0 = {0.f,0.f,0.f,0.f,0.f,0.f,0.f,0.f};
  v8f acc1 = acc0, acc2 = acc0, acc3 = acc0;

  // A staging: one 16B lane-DMA per thread per k-step (bf16, no conversion needed)
  const int am = t >> 2, ak = (t & 3) * 8;
  const unsigned ldsA = (unsigned)(unsigned long long)&lA[am * 32 + ak];
  unsigned* lBu = (unsigned*)lB;

  for (int k0 = kBeg; k0 < kEnd; k0 += 32) {
    // ---- async copy A[64x32] bf16 global -> LDS (ASYNCcnt-tracked, no VGPR round-trip)
    const __bf16* gA = A + (size_t)am * ldA + (k0 + ak);
    asm volatile("global_load_async_to_lds_b128 %0, %1, off"
                 :: "v"(ldsA), "v"(gA) : "memory");
    // ---- W tile 32x128 f32: b128 coalesced loads of two k-adjacent rows,
    //      pack bf16 pairs along k, b32 LDS stores into [n][k] layout
    #pragma unroll
    for (int j = 0; j < 2; ++j) {
      int q  = t + 256 * j;
      int k2 = q >> 5;                 // k pair index 0..15 -> rows 2*k2, 2*k2+1
      int nq = (q & 31) * 4;           // n granule 0..124
      const float4 w0 = *(const float4*)(W + (size_t)(k0 + 2 * k2    ) * N + n0 + nq);
      const float4 w1 = *(const float4*)(W + (size_t)(k0 + 2 * k2 + 1) * N + n0 + nq);
      lBu[(nq + 0) * 16 + k2] = pack_bf16x2(w0.x, w1.x);
      lBu[(nq + 1) * 16 + k2] = pack_bf16x2(w0.y, w1.y);
      lBu[(nq + 2) * 16 + k2] = pack_bf16x2(w0.z, w1.z);
      lBu[(nq + 3) * 16 + k2] = pack_bf16x2(w0.w, w1.w);
    }
    asm volatile("s_wait_asynccnt 0" ::: "memory");
    __syncthreads();
    // ---- issue all fragment loads first, then 4 back-to-back WMMAs
    v16bf af  = ldfrag(&lA[(mBase + lr) * 32 + hl * 8]);
    v16bf bf0 = ldfrag(&lB[(nBase +  0 + lr) * 32 + hl * 8]);
    v16bf bf1 = ldfrag(&lB[(nBase + 32 + lr) * 32 + hl * 8]);
    v16bf bf2 = ldfrag(&lB[(nBase + 64 + lr) * 32 + hl * 8]);
    v16bf bf3 = ldfrag(&lB[(nBase + 96 + lr) * 32 + hl * 8]);
    acc0 = __builtin_amdgcn_wmma_f32_16x16x32_bf16(false, af, false, bf0, (short)0, acc0, false, false);
    acc1 = __builtin_amdgcn_wmma_f32_16x16x32_bf16(false, af, false, bf1, (short)0, acc1, false, false);
    acc2 = __builtin_amdgcn_wmma_f32_16x16x32_bf16(false, af, false, bf2, (short)0, acc2, false, false);
    acc3 = __builtin_amdgcn_wmma_f32_16x16x32_bf16(false, af, false, bf3, (short)0, acc3, false, false);
    __syncthreads();
  }
  // ---- split-K reduce into L2-resident f32 accumulator (one base addr per wave)
  float* base = Cacc + (size_t)(mBase + 8 * hl) * N + n0 + nBase + lr;
  #pragma unroll
  for (int j = 0; j < 8; ++j) {
    float* rp = base + (size_t)j * N;
    atomicAdd(rp +  0, acc0[j]);
    atomicAdd(rp + 32, acc1[j]);
    atomicAdd(rp + 64, acc2[j]);
    atomicAdd(rp + 96, acc3[j]);
  }
}

// ---------------- epilogue: v = relu(acc + bias) (+ add); write f32/f32-dup/bf16 ----------------
__global__ void k_epilogue(const float* __restrict__ Cacc, const float* __restrict__ bias,
                           const float* __restrict__ add, float* __restrict__ outF,
                           float* __restrict__ outF2, __bf16* __restrict__ outBf,
                           int total, int N) {
  int i = blockIdx.x * blockDim.x + threadIdx.x;
  if (i >= total) return;
  int n = i % N;
  float v = Cacc[i] + bias[n];
  v = v > 0.f ? v : 0.f;
  if (add)   v += add[i];
  if (outF)  outF[i]  = v;
  if (outF2) outF2[i] = v;
  if (outBf) outBf[i] = f2bf(v);
}

// =====================================================================================
extern "C" void kernel_launch(void* const* d_in, const int* in_sizes, int n_in,
                              void* d_out, int out_size, void* d_ws, size_t ws_size,
                              hipStream_t stream) {
  const int*   E     = (const int*)  d_in[0];
  const float* Xd    = (const float*)d_in[1];
  const float* cls   = (const float*)d_in[2];
  const float* Wg1   = (const float*)d_in[3];
  const float* Wg2   = (const float*)d_in[4];
  const float* Wbox  = (const float*)d_in[5];
  const float* bbox  = (const float*)d_in[6];
  const float* Wlbl  = (const float*)d_in[7];
  const float* blbl  = (const float*)d_in[8];
  const float* Wd1   = (const float*)d_in[9];
  const float* bd1   = (const float*)d_in[10];
  const float* Wd2   = (const float*)d_in[11];
  const float* bd2   = (const float*)d_in[12];
  const float* Wd3   = (const float*)d_in[13];
  const float* bd3   = (const float*)d_in[14];
  const float* Wz    = (const float*)d_in[15];
  const float* bz    = (const float*)d_in[16];
  float* out = (float*)d_out;

  // ---- workspace layout (bytes), total ~11.5 MB ----
  char* ws = (char*)d_ws;
  size_t oY   = 0;                                    // Adj@X        [N,8]  f32
  size_t oZ   = oY   + (size_t)NTOT * 8 * 4;          // Adj^2@X      [N,8]  f32
  size_t oWg  = oZ   + (size_t)NTOT * 8 * 4;          // fused Wg     [8,16] f32
  size_t oA2  = oWg  + 1024;                          // concat input [64,16544] bf16
  size_t oM   = oA2  + (size_t)BATCH * K2DIM * 2;     // mix_in       [64,16384] bf16
  size_t oMix = oM   + (size_t)BATCH * K1DIM * 2;     // mix          [64,4096]  f32
  size_t oC   = oMix + (size_t)BATCH * H3DIM * 4;     // GEMM accum   [64,4096]  f32
  size_t oH1  = oC   + (size_t)BATCH * H3DIM * 4;     // h after d2   [64,4096]  bf16
  size_t oH2  = oH1  + (size_t)BATCH * H3DIM * 2;     // h after d3   [64,4096]  bf16
  size_t oH3  = oH2  + (size_t)BATCH * H3DIM * 2;     // h after d3'  [64,4096]  bf16

  float*  Y    = (float*)(ws + oY);
  float*  Z    = (float*)(ws + oZ);
  float*  Wg   = (float*)(ws + oWg);
  __bf16* A2   = (__bf16*)(ws + oA2);
  __bf16* Mbf  = (__bf16*)(ws + oM);
  float*  mixf = (float*)(ws + oMix);
  float*  Cacc = (float*)(ws + oC);
  __bf16* H1b  = (__bf16*)(ws + oH1);
  __bf16* H2b  = (__bf16*)(ws + oH2);
  __bf16* H3b  = (__bf16*)(ws + oH3);

  const int* src = E;
  const int* dst = E + NEDGE;

  // ---- GCN part via linearity: x2 = (Adj^2 X) @ (Wg1 @ Wg2) ----
  k_zero_f32<<<2048, 256, 0, stream>>>(Y, NTOT * 8 * 2);            // zero Y and Z (contiguous)
  k_scatter8<<<NEDGE / 256, 256, 0, stream>>>(src, dst, Xd, Y);
  k_scatter8<<<NEDGE / 256, 256, 0, stream>>>(src, dst, Y, Z);
  k_fuse_wg<<<1, 128, 0, stream>>>(Wg1, Wg2, Wg);
  k_build_x2cat<<<NTOT / 256, 256, 0, stream>>>(Z, Wg, A2);
  k_fill_cls<<<(BATCH * NCLS + 255) / 256, 256, 0, stream>>>(cls, A2);

  // ---- box/label branch ----
  k_mixin<<<NTOT / 256, 256, 0, stream>>>(Xd, Wbox, bbox, Wlbl, blbl, Mbf);

  const int total = BATCH * H3DIM;      // 262144
  const int eBlk  = (total + 255) / 256;

  // ---- d1: mix = relu(Mbf @ W_d1 + b_d1) ----  K=16384, split 32 -> 1024 blocks
  k_zero_f32<<<eBlk, 256, 0, stream>>>(Cacc, total);
  k_gemm_bf16<<<dim3(H3DIM / 128, 32), 256, 0, stream>>>(Mbf, K1DIM, Wd1, H3DIM, Cacc, K1DIM, 32);
  k_epilogue<<<eBlk, 256, 0, stream>>>(Cacc, bd1, nullptr, mixf, nullptr, nullptr, total, H3DIM);

  // ---- d2: h1 = relu(A2 @ W_d2 + b_d2) + mix ----  K=16544=32*11*47, split 11 -> 352 blocks
  k_zero_f32<<<eBlk, 256, 0, stream>>>(Cacc, total);
  k_gemm_bf16<<<dim3(H3DIM / 128, 11), 256, 0, stream>>>(A2, K2DIM, Wd2, H3DIM, Cacc, K2DIM, 11);
  k_epilogue<<<eBlk, 256, 0, stream>>>(Cacc, bd2, mixf, nullptr, nullptr, H1b, total, H3DIM);

  // ---- d3 (x2): h = relu(h @ W_d3 + b_d3) ----  K=4096, split 16 -> 512 blocks
  k_zero_f32<<<eBlk, 256, 0, stream>>>(Cacc, total);
  k_gemm_bf16<<<dim3(H3DIM / 128, 16), 256, 0, stream>>>(H1b, H3DIM, Wd3, H3DIM, Cacc, H3DIM, 16);
  k_epilogue<<<eBlk, 256, 0, stream>>>(Cacc, bd3, nullptr, nullptr, nullptr, H2b, total, H3DIM);

  k_zero_f32<<<eBlk, 256, 0, stream>>>(Cacc, total);
  k_gemm_bf16<<<dim3(H3DIM / 128, 16), 256, 0, stream>>>(H2b, H3DIM, Wd3, H3DIM, Cacc, H3DIM, 16);
  k_epilogue<<<eBlk, 256, 0, stream>>>(Cacc, bd3, nullptr, nullptr, nullptr, H3b, total, H3DIM);

  // ---- z: z_mean = z_logvar = relu(h @ W_z + b_z) ----  K=4096, N=512, split 16 -> 64 blocks
  const int ztot = BATCH * LATDIM;      // 32768
  k_zero_f32<<<(ztot + 255) / 256, 256, 0, stream>>>(Cacc, ztot);
  k_gemm_bf16<<<dim3(LATDIM / 128, 16), 256, 0, stream>>>(H3b, H3DIM, Wz, LATDIM, Cacc, H3DIM, 16);
  k_epilogue<<<(ztot + 255) / 256, 256, 0, stream>>>(Cacc, bz, nullptr, out, out + ztot, nullptr,
                                                     ztot, LATDIM);
}
